// Attention_87729001988509
// MI455X (gfx1250) — compile-verified, matrix operand order checked by reading, not run on hardware
//
#include <hip/hip_runtime.h>

// ---------------- problem constants ----------------
#define B_      16
#define S_      2048
#define F_      800         // true feature dim (pad rows of W multiply zeros -> skip)
#define E_      200
#define EPAD    224         // 7*32, contraction padding for q/k/vT (cols >=200 are zero)
#define KSTEPS_F (F_/32)    // 25 contraction steps for projection
#define ESTEPS  (EPAD/32)   // 7 contraction steps over embedding
#define NTILES_E 14         // EPAD/16 projection col tiles
#define NTILES_O 13         // ceil(200/16) output col tiles in attention
#define ROWS    (B_*S_)     // 32768

typedef __bf16 bf16_t;
typedef __attribute__((ext_vector_type(16))) __bf16 v16bf;
typedef __attribute__((ext_vector_type(8)))  float  v8f;

union FragU { v16bf v; uint4 u[2]; };
union FragB { v16bf v; bf16_t h[16]; };

__device__ __forceinline__ v8f wmma_bf16(v16bf a, v16bf b, v8f c) {
  // D = A(16x32 bf16) * B(32x16 bf16) + C(16x16 f32)
  return __builtin_amdgcn_wmma_f32_16x16x32_bf16(false, a, false, b, (short)0, c,
                                                 false, false);
}

// 16-byte async DMA: global -> LDS, tracked by ASYNCcnt.
// LDS address = low 32 bits of the flat shared pointer (wave-relative per ISA
// flat-aperture mapping).
__device__ __forceinline__ void async_cp16(void* lds_dst, const void* gsrc) {
  unsigned lds_off = (unsigned)(size_t)lds_dst;
  unsigned long long ga = (unsigned long long)(size_t)gsrc;
  asm volatile("global_load_async_to_lds_b128 %0, %1, off"
               :: "v"(lds_off), "v"(ga) : "memory");
}
__device__ __forceinline__ void async_wait0() {
  asm volatile("s_wait_asynccnt 0" ::: "memory");
}

#define SOFTMAX_SCALE 0.032274861218395140f   // 1/sqrt(960), as in reference

// ---------------------------------------------------------------------------
// Kernel 1: pack W{q,k,v} (f32 [960,200], only rows <800 matter) into bf16
// B-fragment order: Wb[mat][coltile][kstep][lane][16 elems]. Cols >= 200 -> 0
// so padded activation columns come out exactly zero.
// ---------------------------------------------------------------------------
__global__ void pack_weights_kernel(const float* __restrict__ Wq,
                                    const float* __restrict__ Wk,
                                    const float* __restrict__ Wv,
                                    bf16_t* __restrict__ Wb) {
  int idx = blockIdx.x * blockDim.x + threadIdx.x;     // (mat,ct,step,lane)
  const int total = 3 * NTILES_E * KSTEPS_F * 32;
  if (idx >= total) return;
  int lane = idx & 31;
  int t    = idx >> 5;
  int step = t % KSTEPS_F;  t /= KSTEPS_F;
  int ct   = t % NTILES_E;
  int mat  = t / NTILES_E;
  const float* W = (mat == 0) ? Wq : (mat == 1) ? Wk : Wv;
  int n    = ct * 16 + (lane & 15);
  int half = lane >> 4;
  bf16_t* dst = Wb + (size_t)idx * 16;
  #pragma unroll
  for (int e = 0; e < 16; ++e) {
    int j  = e >> 1;
    int kk = ((j < 4) ? 0 : 16) + half * 8 + (j & 3) * 2 + (e & 1);
    int kg = step * 32 + kk;                    // < 800 always
    float v = (n < E_) ? W[(size_t)kg * E_ + n] : 0.0f;
    dst[e] = (bf16_t)v;
  }
}

// ---------------------------------------------------------------------------
// Kernel 2: fused QKV projection + exact GELU.
// grid = (ROWS/128, 14 coltiles, 3 matrices), block = 256 (8 waves).
// x tile (128x32 fp32) is DMA'd into double-buffered LDS with async-to-LDS;
// fp32->bf16 via native packed converts in the A-fragment build. Weights come
// pre-packed from Wb. q (pre-scaled by 1/sqrt(960)) and k stored row-major
// [row][EPAD]; v stored transposed vT[b][EPAD][S].
// ---------------------------------------------------------------------------
__global__ __launch_bounds__(256) void proj_kernel(
    const float* __restrict__ x, const bf16_t* __restrict__ Wb,
    bf16_t* __restrict__ q, bf16_t* __restrict__ k,
    bf16_t* __restrict__ vT) {
  __shared__ __align__(16) float xs[2][128 * 32];   // 2 x 16 KB fp32 tiles

  const int tid  = threadIdx.x;
  const int wave = tid >> 5;
  const int lane = tid & 31;
  const int half = lane >> 4;
  const int l15  = lane & 15;
  const int rowBase = blockIdx.x * 128;
  const int ct   = blockIdx.y;
  const int mat  = blockIdx.z;

  const float* xrow = x + (size_t)rowBase * F_;
  const bf16_t* wb = Wb + ((size_t)(mat * NTILES_E + ct) * KSTEPS_F) * 32 * 16;

  // stage one 128x32 fp32 tile: 1024 16B chunks / 256 threads = 4 each
  auto stage = [&](int s, float* dst) {
    #pragma unroll
    for (int j = 0; j < 4; ++j) {
      int c  = tid + j * 256;
      int r  = c >> 3;                     // 0..127
      int q4 = c & 7;                      // float4 within 32-wide row
      async_cp16(dst + r * 32 + q4 * 4,
                 xrow + (size_t)r * F_ + s * 32 + q4 * 4);
    }
  };

  stage(0, xs[0]);
  async_wait0();
  __syncthreads();

  v8f acc = {};
  for (int s = 0; s < KSTEPS_F; ++s) {
    int cur = s & 1;
    if (s + 1 < KSTEPS_F) stage(s + 1, xs[cur ^ 1]);

    // A frag: 4 x ds_load_b128 of fp32, native packed cvt to bf16
    const float* base = xs[cur] + (wave * 16 + l15) * 32 + half * 8;
    float4 f0 = *(const float4*)(base);
    float4 f1 = *(const float4*)(base + 4);
    float4 f2 = *(const float4*)(base + 16);
    float4 f3 = *(const float4*)(base + 20);
    FragB au;
    au.h[0]  = (bf16_t)f0.x; au.h[1]  = (bf16_t)f0.y;
    au.h[2]  = (bf16_t)f0.z; au.h[3]  = (bf16_t)f0.w;
    au.h[4]  = (bf16_t)f1.x; au.h[5]  = (bf16_t)f1.y;
    au.h[6]  = (bf16_t)f1.z; au.h[7]  = (bf16_t)f1.w;
    au.h[8]  = (bf16_t)f2.x; au.h[9]  = (bf16_t)f2.y;
    au.h[10] = (bf16_t)f2.z; au.h[11] = (bf16_t)f2.w;
    au.h[12] = (bf16_t)f3.x; au.h[13] = (bf16_t)f3.y;
    au.h[14] = (bf16_t)f3.z; au.h[15] = (bf16_t)f3.w;

    // B frag from pre-packed weights: 2 x coalesced global b128 per lane
    FragU bu;
    const bf16_t* wsrc = wb + ((size_t)s * 32 + lane) * 16;
    bu.u[0] = *(const uint4*)(wsrc);
    bu.u[1] = *(const uint4*)(wsrc + 8);

    acc = wmma_bf16(au.v, bu.v, acc);

    async_wait0();      // next tile's DMA done (own lanes)
    __syncthreads();    // ... and everyone else's
  }

  // epilogue: exact GELU, store bf16 (q pre-scaled for softmax)
  const float inv_sqrt2 = 0.7071067811865476f;
  const int n = ct * 16 + l15;
  #pragma unroll
  for (int r8 = 0; r8 < 8; ++r8) {
    float a = acc[r8];
    float g = 0.5f * a * (1.0f + erff(a * inv_sqrt2));
    int row = rowBase + wave * 16 + r8 + half * 8;
    if (mat == 0) {
      q[(size_t)row * EPAD + n] = (bf16_t)(g * SOFTMAX_SCALE);
    } else if (mat == 1) {
      k[(size_t)row * EPAD + n] = (bf16_t)g;
    } else {
      int b = row >> 11, srow = row & (S_ - 1);
      vT[((size_t)b * EPAD + n) * S_ + srow] = (bf16_t)g;
    }
  }
}

// ---------------------------------------------------------------------------
// Kernel 3: flash attention with block-shared, double-buffered async K/V
// staging. 8 waves of a block handle 8 consecutive 16-query tiles of the SAME
// batch, so one DMA of each 32-key K tile (14 KB, fully contiguous) + V tile
// (208x32, 64 B strips) serves all 8 waves -> 8x less L2 fragment traffic.
// Per key-block per wave: 14 WMMAs (QK^T) + online softmax + 13 WMMAs (P.V).
// ---------------------------------------------------------------------------
#define KCHUNKS (32 * EPAD * 2 / 16)        // 896
#define VCHUNKS (208 * 32 * 2 / 16)         // 832
#define TOTCHUNKS (KCHUNKS + VCHUNKS)       // 1728

__global__ __launch_bounds__(256) void attn_kernel(
    const bf16_t* __restrict__ q, const bf16_t* __restrict__ k,
    const bf16_t* __restrict__ vT, float* __restrict__ out) {
  __shared__ __align__(16) bf16_t ksb[2][32 * EPAD];   // 2 x 14 KB
  __shared__ __align__(16) bf16_t vsb[2][208 * 32];    // 2 x 13 KB
  __shared__ __align__(16) bf16_t pbuf[8][16 * 32];    // P staging

  const int tid  = threadIdx.x;
  const int wave = tid >> 5;
  const int lane = tid & 31;
  const int half = lane >> 4;
  const int l15  = lane & 15;

  const int job  = blockIdx.x * 8 + wave;   // 0..2047
  const int b    = job >> 7;                // batch (same for all 8 waves)
  const int qt   = job & 127;               // query tile in batch
  const int qrow0 = b * S_ + qt * 16;

  const bf16_t* kbase = k  + (size_t)b * S_ * EPAD;
  const bf16_t* vbase = vT + (size_t)b * EPAD * S_;

  // stage K(32xEPAD) + V(208x32) tiles for key-block kb via async DMA
  auto stage_kv = [&](int kb, bf16_t* ks, bf16_t* vs) {
    const bf16_t* ksrc = kbase + (size_t)kb * EPAD;  // contiguous 14 KB span
    #pragma unroll
    for (int j = 0; j < 7; ++j) {
      int c = tid + j * 256;
      if (c < KCHUNKS) {
        async_cp16(ks + c * 8, ksrc + (size_t)c * 8);
      } else if (c < TOTCHUNKS) {
        int cc = c - KCHUNKS;
        int row = cc >> 2, part = cc & 3;                    // 4 chunks / row
        async_cp16(vs + row * 32 + part * 8,
                   vbase + (size_t)row * S_ + kb + part * 8);
      }
    }
  };

  // Q fragments (pre-scaled by 1/sqrt(960) at projection) kept in registers
  v16bf qf[ESTEPS];
  {
    const bf16_t* qb = q + (size_t)(qrow0 + l15) * EPAD + half * 8;
    #pragma unroll
    for (int s = 0; s < ESTEPS; ++s) {
      FragU au;
      au.u[0] = *(const uint4*)(qb + s * 32);
      au.u[1] = *(const uint4*)(qb + s * 32 + 16);
      qf[s] = au.v;
    }
  }

  v8f o[NTILES_O];
  const v8f vzero = {};
  #pragma unroll
  for (int t = 0; t < NTILES_O; ++t) o[t] = vzero;
  float mstat[8], lstat[8];
  #pragma unroll
  for (int r = 0; r < 8; ++r) { mstat[r] = -1e30f; lstat[r] = 0.0f; }

  bf16_t* pl = &pbuf[wave][0];

  stage_kv(0, ksb[0], vsb[0]);
  async_wait0();
  __syncthreads();

  for (int it = 0; it < S_ / 32; ++it) {
    const int cur = it & 1;
    if (it + 1 < S_ / 32) stage_kv((it + 1) * 32, ksb[cur ^ 1], vsb[cur ^ 1]);
    const bf16_t* ks = ksb[cur];
    const bf16_t* vs = vsb[cur];

    // ---- scores: two 16-key tiles over 7 e-steps (frags from LDS) -------
    v8f s0 = vzero, s1 = vzero;
    const bf16_t* k0 = ks + (size_t)l15 * EPAD + half * 8;
    const bf16_t* k1 = k0 + 16 * EPAD;
    #pragma unroll
    for (int s = 0; s < ESTEPS; ++s) {
      FragU bu;
      bu.u[0] = *(const uint4*)(k0 + s * 32);
      bu.u[1] = *(const uint4*)(k0 + s * 32 + 16);
      s0 = wmma_bf16(qf[s], bu.v, s0);
      bu.u[0] = *(const uint4*)(k1 + s * 32);
      bu.u[1] = *(const uint4*)(k1 + s * 32 + 16);
      s1 = wmma_bf16(qf[s], bu.v, s1);
    }

    // ---- online softmax (row stats live per lane, rows = reg + half*8) --
    v8f corrv;
    #pragma unroll
    for (int r = 0; r < 8; ++r) {
      float a0 = s0[r], a1 = s1[r];       // scale already folded into q
      float mx = fmaxf(a0, a1);
      #pragma unroll
      for (int off = 1; off < 16; off <<= 1)
        mx = fmaxf(mx, __shfl_xor(mx, off, 32));   // stays inside 16-lane half
      float mnew = fmaxf(mstat[r], mx);
      float c = __expf(mstat[r] - mnew);
      corrv[r] = c;
      mstat[r] = mnew;
      a0 = __expf(a0 - mnew);
      a1 = __expf(a1 - mnew);
      s0[r] = a0; s1[r] = a1;
      float rs = a0 + a1;
      #pragma unroll
      for (int off = 1; off < 16; off <<= 1)
        rs += __shfl_xor(rs, off, 32);
      lstat[r] = lstat[r] * c + rs;
    }

    // ---- P: C-layout (lane=key col) -> A-layout (lane=q row) via LDS ----
    #pragma unroll
    for (int r = 0; r < 8; ++r) {
      int m = r + half * 8;
      pl[m * 32 + l15]      = (bf16_t)s0[r];
      pl[m * 32 + 16 + l15] = (bf16_t)s1[r];
    }
    asm volatile("s_wait_dscnt 0" ::: "memory");   // per-wave LDS RAW fence
    FragU pu;
    {
      const bf16_t* base = pl + l15 * 32 + half * 8;
      pu.u[0] = *(const uint4*)(base);
      pu.u[1] = *(const uint4*)(base + 16);
    }

    // ---- O = O*corr + P @ V (V strips contiguous along keys in LDS) -----
    #pragma unroll
    for (int t = 0; t < NTILES_O; ++t) {
      o[t] = o[t] * corrv;                 // vector mul -> packed f32 ops
      const bf16_t* vp = vs + (size_t)(t * 16 + l15) * 32 + half * 8;
      FragU bu;
      bu.u[0] = *(const uint4*)(vp);
      bu.u[1] = *(const uint4*)(vp + 16);
      o[t] = wmma_bf16(pu.v, bu.v, o[t]);
    }

    async_wait0();      // next tile's DMA done (own lanes)
    __syncthreads();    // ... and all other lanes' DMA
  }

  // ---- epilogue: 8 reciprocals, then multiply + coalesced fp32 stores ---
  float linv[8];
  #pragma unroll
  for (int r = 0; r < 8; ++r) linv[r] = 1.0f / lstat[r];
  #pragma unroll
  for (int t = 0; t < NTILES_O; ++t) {
    int e = t * 16 + l15;
    if (e < E_) {
      #pragma unroll
      for (int r = 0; r < 8; ++r) {
        int qrow = qt * 16 + r + half * 8;
        out[((size_t)b * S_ + qrow) * E_ + e] = o[t][r] * linv[r];
      }
    }
  }
}

// ---------------------------------------------------------------------------
// launcher
// ---------------------------------------------------------------------------
extern "C" void kernel_launch(void* const* d_in, const int* in_sizes, int n_in,
                              void* d_out, int out_size, void* d_ws, size_t ws_size,
                              hipStream_t stream) {
  const float* x  = (const float*)d_in[0];
  const float* Wq = (const float*)d_in[1];
  const float* Wk = (const float*)d_in[2];
  const float* Wv = (const float*)d_in[3];
  float* out = (float*)d_out;

  char* ws = (char*)d_ws;
  bf16_t* Wb = (bf16_t*)(ws);                            // 1.05 MB
  bf16_t* qb = (bf16_t*)(ws + (size_t)2  * 1024 * 1024); // 14.7 MB
  bf16_t* kb = (bf16_t*)(ws + (size_t)18 * 1024 * 1024); // 14.7 MB
  bf16_t* vb = (bf16_t*)(ws + (size_t)34 * 1024 * 1024); // 14.7 MB

  // 1) pack weights into WMMA B-fragment order (bf16, padded cols zeroed)
  {
    int total = 3 * NTILES_E * KSTEPS_F * 32;
    pack_weights_kernel<<<(total + 255) / 256, 256, 0, stream>>>(Wq, Wk, Wv, Wb);
  }
  // 2) fused QKV projection + GELU (q pre-scaled by 1/sqrt(960))
  {
    dim3 grid(ROWS / 128, NTILES_E, 3);
    proj_kernel<<<grid, 256, 0, stream>>>(x, Wb, qb, kb, vb);
  }
  // 3) flash attention
  attn_kernel<<<256, 256, 0, stream>>>(qb, kb, vb, out);
}
